// MHSAttention_8976481648722
// MI455X (gfx1250) — matrix operand order repaired and checked by reading, not compiled, not run on hardware
//
#include <hip/hip_runtime.h>
#include <cstdint>
#include <cstddef>

typedef __attribute__((ext_vector_type(16))) __bf16 v16bf;
typedef __attribute__((ext_vector_type(8)))  float  v8f;

static __device__ __forceinline__ v8f wmma_bf16(v16bf a, v16bf b, v8f c) {
  // D = A(16x32 bf16) * B(32x16 bf16) + C(16x16 f32)
  return __builtin_amdgcn_wmma_f32_16x16x32_bf16(false, a, false, b, (short)0, c, false, false);
}

// CDNA5 async global->LDS copy (ASYNCcnt-tracked DMA path).
// LDS address = low 32 bits of the flat pointer (LDS aperture truncation, ISA 10.2).
static __device__ __forceinline__ void async_copy_b128(const __bf16* gptr, __bf16* lptr) {
  unsigned lds = (unsigned)(uintptr_t)lptr;
  asm volatile("global_load_async_to_lds_b128 %0, %1, off"
               :: "v"(lds), "v"((unsigned long long)(uintptr_t)gptr) : "memory");
}
static __device__ __forceinline__ void wait_async() {
  asm volatile("s_wait_asynccnt 0x0" ::: "memory");
}

constexpr int DIM   = 768;
constexpr int NH    = 12;
constexpr int HD    = 64;     // head dim
constexpr int NTOK  = 1024;   // 32*32 tokens per image
constexpr int BATCH = 8;
constexpr int QLD   = 3 * DIM;        // 2304, row stride of fused qkv buffer
constexpr int MROWS = BATCH * NTOK;   // 8192

// ---------------- f32 -> bf16 conversion ----------------
__global__ __launch_bounds__(256)
void cvt_f32_to_bf16(const float* __restrict__ in, __bf16* __restrict__ out, int n4) {
  int i = blockIdx.x * 256 + threadIdx.x;
  if (i < n4) {
    float4 v = reinterpret_cast<const float4*>(in)[i];
    __bf16* o = out + (size_t)i * 4;
    o[0] = (__bf16)v.x; o[1] = (__bf16)v.y; o[2] = (__bf16)v.z; o[3] = (__bf16)v.w;
  }
}

// ---------------- 128x256 block-tile WMMA GEMM ----------------
// C[M,N] = A[M,K](bf16) * B[K,N](bf16); BF16OUT ? bf16 store : f32 + bias.
// 256 threads = 8 waves in a 2x4 grid; wave tile 64x64 = 4x4 WMMA tiles.
// A staged via async DMA to LDS; B staged transposed (N-major) for contiguous
// B-fragment reads.
template<bool BF16OUT>
__global__ __launch_bounds__(256)
void wmma_gemm256(const __bf16* __restrict__ A, const __bf16* __restrict__ B,
                  const float* __restrict__ bias, void* __restrict__ Cout,
                  int M, int N, int K) {
  constexpr int LDT = 72;                 // 64 + 8 pad (144B rows, 16B aligned)
  __shared__ __bf16 sA[128 * LDT];        // A tile, row-major [m][k]   (18KB)
  __shared__ __bf16 sBt[256 * LDT];       // B tile, N-major  [n][k]   (36KB)
  const int tid  = threadIdx.x;
  const int lane = tid & 31, wave = tid >> 5;
  const int wm = (wave >> 2) * 64, wn = (wave & 3) * 64;
  const int row0 = blockIdx.y * 128, col0 = blockIdx.x * 256;
  const int r = lane & 15, hl = lane >> 4;
  const int kh8 = hl * 8, kh16 = hl * 16;

  v8f acc[4][4];
#pragma unroll
  for (int mt = 0; mt < 4; ++mt)
#pragma unroll
    for (int nt = 0; nt < 4; ++nt) acc[mt][nt] = 0.0f;

  for (int k0 = 0; k0 < K; k0 += 64) {
    // --- stage A (128x64): 1024 x 16B chunks via async global->LDS DMA ---
#pragma unroll
    for (int j = 0; j < 4; ++j) {
      const int c = j * 256 + tid;          // consecutive lanes -> consecutive 16B
      const int arow = c >> 3;              // 8 chunks per 64-elem row
      const int ac = (c & 7) * 8;
      async_copy_b128(A + (size_t)(row0 + arow) * K + k0 + ac, &sA[arow * LDT + ac]);
    }
    // --- stage B (64x256): coalesced read, transposed scatter into sBt[n][k] ---
    {
      const int brow = tid >> 2;            // 0..63
      const int bc = (tid & 3) * 64;        // 0,64,128,192
      union { uint4 u[8]; __bf16 h[64]; } tb;
      const uint4* gp = reinterpret_cast<const uint4*>(B + (size_t)(k0 + brow) * N + col0 + bc);
#pragma unroll
      for (int i = 0; i < 8; ++i) tb.u[i] = gp[i];
#pragma unroll
      for (int i = 0; i < 64; ++i) sBt[(bc + i) * LDT + brow] = tb.h[i];
    }
    wait_async();
    __syncthreads();

#pragma unroll
    for (int ks = 0; ks < 2; ++ks) {
      v16bf af[4], bf[4];
#pragma unroll
      for (int mt = 0; mt < 4; ++mt) {
        // ISA 16-bit A layout: lane -> row, K(e) = 16*(e>=8) + 8*(lane>=16) + (e&7)
        const __bf16* p = &sA[(wm + mt * 16 + r) * LDT + ks * 32 + kh8];
#pragma unroll
        for (int e = 0; e < 16; ++e) af[mt][e] = p[((e >> 3) << 4) + (e & 7)];
      }
#pragma unroll
      for (int nt = 0; nt < 4; ++nt) {
        // ISA 16-bit B layout: lane -> col, K(e) = 16*(e>=8) + 8*(lane>=16) + (e&7)
        const __bf16* p = &sBt[(wn + nt * 16 + r) * LDT + ks * 32 + kh8];
#pragma unroll
        for (int e = 0; e < 16; ++e) bf[nt][e] = p[((e >> 3) << 4) + (e & 7)];
      }
#pragma unroll
      for (int mt = 0; mt < 4; ++mt)
#pragma unroll
        for (int nt = 0; nt < 4; ++nt)
          acc[mt][nt] = wmma_bf16(af[mt], bf[nt], acc[mt][nt]);
    }
    __syncthreads();
  }

  // C/D layout: VGPR slot i -> row i + 8*(lane>=16), col = lane&15
#pragma unroll
  for (int mt = 0; mt < 4; ++mt)
#pragma unroll
    for (int nt = 0; nt < 4; ++nt)
#pragma unroll
      for (int i = 0; i < 8; ++i) {
        const int row = row0 + wm + mt * 16 + i + hl * 8;
        const int col = col0 + wn + nt * 16 + r;
        const float v = acc[mt][nt][i];
        if (BF16OUT) reinterpret_cast<__bf16*>(Cout)[(size_t)row * N + col] = (__bf16)v;
        else         reinterpret_cast<float*>(Cout)[(size_t)row * N + col]  = v + bias[col];
      }
}

// ---------------- flash-attention over fused qkv buffer ----------------
// qkv: [MROWS][2304] bf16 (q | k | v per row). aout: [MROWS][768] bf16.
// block = (b, head, 128-query tile); wave owns 16 query rows; key tiles of 64.
__global__ __launch_bounds__(256)
void attn_kernel(const __bf16* __restrict__ qkv, __bf16* __restrict__ aout) {
  constexpr int LDT = 72;
  __shared__ __bf16 sK[64 * LDT];        // [key][d]
  __shared__ __bf16 sVt[64 * LDT];       // [d][key]  (N-major for P@V)
  __shared__ __bf16 sP[8 * 16 * LDT];    // per-wave 16 x 64 P tile

  const int blk  = blockIdx.x;
  const int qt   = blk & 7;
  const int head = (blk >> 3) % NH;
  const int b    = blk / (8 * NH);
  const int tid = threadIdx.x, lane = tid & 31, wave = tid >> 5;
  const int r = lane & 15, hl = lane >> 4;
  const int kh8 = hl * 8, kh16 = hl * 16;
  const size_t qrow0 = (size_t)b * NTOK + qt * 128 + wave * 16;
  __bf16* sPw = sP + wave * 16 * LDT;
  const float scale = 0.125f; // hd^-0.5

  // Q A-fragments for the whole key loop: 16 rows x 64, two k=32 slabs
  v16bf qf[2];
  {
    const __bf16* qp = qkv + (qrow0 + r) * QLD + head * HD;
#pragma unroll
    for (int ks = 0; ks < 2; ++ks)
#pragma unroll
      for (int e = 0; e < 16; ++e)
        qf[ks][e] = qp[ks * 32 + kh8 + ((e >> 3) << 4) + (e & 7)];
  }

  float m_i[8], l_i[8];
  v8f oacc[4];
#pragma unroll
  for (int i = 0; i < 8; ++i) { m_i[i] = -1e30f; l_i[i] = 0.0f; }
#pragma unroll
  for (int nt = 0; nt < 4; ++nt) oacc[nt] = 0.0f;

  for (int kt = 0; kt < NTOK / 64; ++kt) {
    __syncthreads();
    { // stage K row-major + V transposed; thread copies 16 contiguous d-values
      const int key = tid >> 2;
      const int d0 = (tid & 3) * 16;
      const __bf16* kp = qkv + ((size_t)b * NTOK + kt * 64 + key) * QLD + DIM + head * HD + d0;
      const uint4* kp4 = reinterpret_cast<const uint4*>(kp);
      uint4* sp4 = reinterpret_cast<uint4*>(&sK[key * LDT + d0]);
      sp4[0] = kp4[0]; sp4[1] = kp4[1];
      union { uint4 u[2]; __bf16 h[16]; } tv;
      const uint4* vp4 = reinterpret_cast<const uint4*>(kp + DIM);
      tv.u[0] = vp4[0]; tv.u[1] = vp4[1];
#pragma unroll
      for (int i = 0; i < 16; ++i) sVt[(d0 + i) * LDT + key] = tv.h[i];
    }
    __syncthreads();

    // S = Q @ K^T : 16 x 64 (4 n-tiles, 2 k-steps each)
    v8f s[4];
#pragma unroll
    for (int nt = 0; nt < 4; ++nt) s[nt] = 0.0f;
#pragma unroll
    for (int ks = 0; ks < 2; ++ks)
#pragma unroll
      for (int nt = 0; nt < 4; ++nt) {
        v16bf bf;
        const __bf16* p = &sK[(nt * 16 + r) * LDT + ks * 32 + kh8];
#pragma unroll
        for (int e = 0; e < 16; ++e) bf[e] = p[((e >> 3) << 4) + (e & 7)];
        s[nt] = wmma_bf16(qf[ks], bf, s[nt]);
      }

    // online softmax: row i+8*hl lives in VGPR slot i across the 16-lane half
    float mnew[8], rs[8], alpha[8];
#pragma unroll
    for (int i = 0; i < 8; ++i) {
      float mv = fmaxf(fmaxf(s[0][i], s[1][i]), fmaxf(s[2][i], s[3][i])) * scale;
#pragma unroll
      for (int mk = 1; mk <= 8; mk <<= 1) mv = fmaxf(mv, __shfl_xor(mv, mk, 32));
      mnew[i]  = fmaxf(mv, m_i[i]);
      alpha[i] = __expf(m_i[i] - mnew[i]);
      rs[i]    = 0.0f;
    }
#pragma unroll
    for (int nt = 0; nt < 4; ++nt)
#pragma unroll
      for (int i = 0; i < 8; ++i) {
        float p = __expf(s[nt][i] * scale - mnew[i]);
        rs[i] += p;
        sPw[(i + hl * 8) * LDT + nt * 16 + r] = (__bf16)p;   // C-layout -> LDS
      }
#pragma unroll
    for (int i = 0; i < 8; ++i) {
#pragma unroll
      for (int mk = 1; mk <= 8; mk <<= 1) rs[i] += __shfl_xor(rs[i], mk, 32);
      l_i[i] = l_i[i] * alpha[i] + rs[i];
      m_i[i] = mnew[i];
    }
#pragma unroll
    for (int nt = 0; nt < 4; ++nt)
#pragma unroll
      for (int i = 0; i < 8; ++i) oacc[nt][i] *= alpha[i];

    // O += P @ V : reload P from LDS as A-fragments, V^T is N-major B
#pragma unroll
    for (int ks = 0; ks < 2; ++ks) {
      v16bf pa;
      const __bf16* pp = &sPw[r * LDT + ks * 32 + kh8];
#pragma unroll
      for (int e = 0; e < 16; ++e) pa[e] = pp[((e >> 3) << 4) + (e & 7)];
#pragma unroll
      for (int nt = 0; nt < 4; ++nt) {
        v16bf vb;
        const __bf16* p = &sVt[(nt * 16 + r) * LDT + ks * 32 + kh8];
#pragma unroll
        for (int e = 0; e < 16; ++e) vb[e] = p[((e >> 3) << 4) + (e & 7)];
        oacc[nt] = wmma_bf16(pa, vb, oacc[nt]);
      }
    }
  }

  // normalize and store into [MROWS][768] bf16 at column block head*64
#pragma unroll
  for (int i = 0; i < 8; ++i) {
    const float inv = 1.0f / l_i[i];
#pragma unroll
    for (int nt = 0; nt < 4; ++nt)
      aout[(qrow0 + i + hl * 8) * DIM + head * HD + nt * 16 + r] =
          (__bf16)(oacc[nt][i] * inv);
  }
}

// ---------------- launch ----------------
extern "C" void kernel_launch(void* const* d_in, const int* in_sizes, int n_in,
                              void* d_out, int out_size, void* d_ws, size_t ws_size,
                              hipStream_t stream) {
  const float* x     = (const float*)d_in[0];   // [8,32,32,768]
  const float* w_qkv = (const float*)d_in[1];   // [768,2304]
  const float* w_out = (const float*)d_in[2];   // [768,768]
  const float* b_out = (const float*)d_in[3];   // [768]
  float* out = (float*)d_out;

  const size_t nX    = (size_t)MROWS * DIM;   // 6291456
  const size_t nWqkv = (size_t)DIM * QLD;     // 1769472
  const size_t nWout = (size_t)DIM * DIM;     // 589824
  const size_t nQKV  = (size_t)MROWS * QLD;
  const size_t nAO   = (size_t)MROWS * DIM;

  char* ws = (char*)d_ws;                       // ~64.5 MB total
  __bf16* xb    = (__bf16*)ws; ws += nX * 2;
  __bf16* wqkvb = (__bf16*)ws; ws += nWqkv * 2;
  __bf16* woutb = (__bf16*)ws; ws += nWout * 2;
  __bf16* qkvb  = (__bf16*)ws; ws += nQKV * 2;
  __bf16* aob   = (__bf16*)ws; ws += nAO * 2;
  (void)ws_size; (void)in_sizes; (void)n_in; (void)out_size;

  cvt_f32_to_bf16<<<(int)(nX    / 4 / 256), 256, 0, stream>>>(x,     xb,    (int)(nX    / 4));
  cvt_f32_to_bf16<<<(int)(nWqkv / 4 / 256), 256, 0, stream>>>(w_qkv, wqkvb, (int)(nWqkv / 4));
  cvt_f32_to_bf16<<<(int)(nWout / 4 / 256), 256, 0, stream>>>(w_out, woutb, (int)(nWout / 4));

  // qkv = x @ w_qkv        [8192 x 2304]
  wmma_gemm256<true><<<dim3(QLD / 256, MROWS / 128), 256, 0, stream>>>(
      xb, wqkvb, nullptr, qkvb, MROWS, QLD, DIM);

  // attention -> [8192 x 768] bf16
  attn_kernel<<<BATCH * NH * (NTOK / 128), 256, 0, stream>>>(qkvb, aob);

  // out = attn_out @ w_out + b_out   [8192 x 768] f32
  wmma_gemm256<false><<<dim3(DIM / 256, MROWS / 128), 256, 0, stream>>>(
      aob, woutb, b_out, out, MROWS, DIM, DIM);
}
